// CausalSelfAttention_2860448219236
// MI455X (gfx1250) — matrix operand order repaired and checked by reading, not compiled, not run on hardware
//
#include <hip/hip_runtime.h>

// ---------------------------------------------------------------------------
// CausalSelfAttention for MI455X (gfx1250, wave32)
//   WMMA f32_16x16x32_f16 for QKV projections, QK^T and PV.
//   Async global->LDS (ASYNCcnt) double-buffered K/V staging per block.
//   B=2, S=2048, H=1024, NH=16, HD=64, WINDOW=512, NEG=-1e4
// Workspace: Q[b,h,s,d], K[b,h,s,d], V^T[b,h,d,s] as f16 (3 x 8 MB)
// ---------------------------------------------------------------------------

typedef __attribute__((ext_vector_type(16))) _Float16 v16h;
typedef __attribute__((ext_vector_type(8)))  _Float16 h8;
typedef __attribute__((ext_vector_type(8)))  float    v8f;
typedef __attribute__((ext_vector_type(4)))  float    f4;

#define B_      2
#define S_      2048
#define H_      1024
#define NH_     16
#define HD_     64
#define WINDOW_ 512
#define NEG_    (-10000.0f)

// padded LDS row strides (halves) to spread banks
#define KROW_   72      // 64 data + 8 pad  (144 B)
#define VROW_   40      // 32 data + 8 pad  ( 80 B)

__device__ __forceinline__ float rowmax16(float x) {
#pragma unroll
  for (int m = 1; m < 16; m <<= 1) x = fmaxf(x, __shfl_xor(x, m, 32));
  return x;
}
__device__ __forceinline__ float rowsum16(float x) {
#pragma unroll
  for (int m = 1; m < 16; m <<= 1) x += __shfl_xor(x, m, 32);
  return x;
}

// 16-byte async DMA global -> LDS, tracked by ASYNCcnt
__device__ __forceinline__ void async_copy16(unsigned lds_addr, const void* gaddr) {
  asm volatile("global_load_async_to_lds_b128 %0, %1, off"
               :: "v"(lds_addr), "v"((unsigned long long)(uintptr_t)gaddr)
               : "memory");
}
__device__ __forceinline__ void wait_async0() {
  asm volatile("s_wait_asynccnt 0x0" ::: "memory");
}

// ---------------------------------------------------------------------------
// Kernel 1: (X @ W^T + bias) -> f16. One wave per 16x64 output tile:
//   one A (X) fragment per k-step feeds 4 WMMAs (4 N-subtiles).
//   VT == 0: out[b,h,s,d]   VT == 1: out[b,h,d,s]  (transposed store for V)
// ---------------------------------------------------------------------------
template <int VT>
__global__ void __launch_bounds__(128)
qkv_gemm_f16(const float* __restrict__ X, const float* __restrict__ W,
             const float* __restrict__ bias, _Float16* __restrict__ out) {
  const unsigned lane  = threadIdx.x & 31u;
  const unsigned wave  = blockIdx.x * (blockDim.x >> 5) + (threadIdx.x >> 5);
  const unsigned tilesN = H_ / 64u;             // 16 (64-wide N tiles)
  const unsigned tileM  = wave / tilesN;        // 0..255  (B*S/16)
  const unsigned tileN  = wave % tilesN;        // 0..15
  const unsigned hs = lane >> 4;                // half-wave select
  const unsigned lr = lane & 15u;

  const float* xrow  = X + (size_t)(tileM * 16u + lr) * H_;        // A: lane=M
  const float* wrow0 = W + (size_t)(tileN * 64u + lr) * H_;        // B: lane=N

  v8f acc[4] = {};
  for (unsigned k0 = 0; k0 < H_; k0 += 32u) {
    // A fragment: elements 0-15 -> K = hs*16+e
    f4 xa0 = *(const f4*)(xrow + k0 + hs * 16u);
    f4 xa1 = *(const f4*)(xrow + k0 + hs * 16u + 4u);
    f4 xb0 = *(const f4*)(xrow + k0 + hs * 16u + 8u);
    f4 xb1 = *(const f4*)(xrow + k0 + hs * 16u + 12u);
    v16h a;
#pragma unroll
    for (int e = 0; e < 4; ++e) {
      a[e]      = (_Float16)xa0[e];
      a[e + 4]  = (_Float16)xa1[e];
      a[e + 8]  = (_Float16)xb0[e];
      a[e + 12] = (_Float16)xb1[e];
    }
#pragma unroll
    for (int n = 0; n < 4; ++n) {
      // B fragment: elements 0-15 -> K = hs*16+e (lane = N holds W[col, k])
      const float* wrow = wrow0 + (size_t)n * 16u * H_;
      f4 w0 = *(const f4*)(wrow + k0 + hs * 16u);
      f4 w1 = *(const f4*)(wrow + k0 + hs * 16u + 4u);
      f4 w2 = *(const f4*)(wrow + k0 + hs * 16u + 8u);
      f4 w3 = *(const f4*)(wrow + k0 + hs * 16u + 12u);
      v16h bf;
#pragma unroll
      for (int e = 0; e < 4; ++e) {
        bf[e]      = (_Float16)w0[e];
        bf[e + 4]  = (_Float16)w1[e];
        bf[e + 8]  = (_Float16)w2[e];
        bf[e + 12] = (_Float16)w3[e];
      }
      acc[n] = __builtin_amdgcn_wmma_f32_16x16x32_f16(false, a, false, bf,
                                                      (short)0, acc[n],
                                                      false, false);
    }
  }

#pragma unroll
  for (int n = 0; n < 4; ++n) {
    const unsigned col = tileN * 64u + (unsigned)n * 16u + lr;  // = h*HD + d
    const float bvv = bias[col];
    const unsigned hh = col >> 6, d = col & 63u;
#pragma unroll
    for (int r = 0; r < 8; ++r) {
      const unsigned m   = (unsigned)r + 8u * hs;
      const unsigned row = tileM * 16u + m;                     // = b*S + s
      const unsigned bb = row >> 11, s = row & (S_ - 1u);
      const size_t idx = VT
          ? (((size_t)bb * NH_ + hh) * HD_ + d) * S_ + s        // V^T: [b,h,d,s]
          : (((size_t)bb * NH_ + hh) * S_ + s) * HD_ + d;       //      [b,h,s,d]
      out[idx] = (_Float16)(acc[n][r] + bvv);
    }
  }
}

// ---------------------------------------------------------------------------
// Kernel 2: flash attention. One block = one (b,h) x 64 query rows (4 waves).
//   Uniform loop over 32-key blocks; K and V^T tiles staged into LDS with
//   double-buffered async copies; online softmax; WMMA for QK^T and PV.
// ---------------------------------------------------------------------------
__global__ void __launch_bounds__(128)
flash_attn(const _Float16* __restrict__ Q, const _Float16* __restrict__ K,
           const _Float16* __restrict__ Vt, const float* __restrict__ amask,
           float* __restrict__ out) {
  __shared__ _Float16 kbuf[2][32 * KROW_];   // 32 keys x 64 d (padded rows)
  __shared__ _Float16 vbuf[2][64 * VROW_];   // 64 d x 32 keys (padded rows)
  __shared__ _Float16 pshare[4][16 * 32];    // per-wave P staging (C -> A)

  const unsigned lane = threadIdx.x & 31u;
  const unsigned wid  = threadIdx.x >> 5;
  const unsigned qBlocks = S_ / 64u;               // 32
  const unsigned bh    = blockIdx.x / qBlocks;
  const unsigned qbase = (blockIdx.x % qBlocks) * 64u;
  const unsigned b  = bh / NH_;
  const unsigned h  = bh % NH_;
  const unsigned hs = lane >> 4;
  const unsigned lr = lane & 15u;
  const unsigned qt = (qbase >> 4) + wid;          // this wave's 16-row q tile

  const _Float16* Qb  = Q  + (size_t)bh * S_ * HD_;
  const _Float16* Kb  = K  + (size_t)bh * S_ * HD_;
  const _Float16* Vtb = Vt + (size_t)bh * HD_ * S_;
  const float*    am  = amask + (size_t)b * S_;
  _Float16* plds = &pshare[wid][0];

  // ---- Q A-fragments (d = 0..31 and 32..63), lane = M row
  const _Float16* qrow = Qb + (size_t)(qt * 16u + lr) * HD_;
  v16h qa0, qa1;
  {
    h8 x0 = *(const h8*)(qrow + hs * 8u);
    h8 x1 = *(const h8*)(qrow + 16u + hs * 8u);
    h8 x2 = *(const h8*)(qrow + 32u + hs * 8u);
    h8 x3 = *(const h8*)(qrow + 48u + hs * 8u);
#pragma unroll
    for (int e = 0; e < 8; ++e) {
      qa0[e] = x0[e]; qa0[e + 8] = x1[e];
      qa1[e] = x2[e]; qa1[e + 8] = x3[e];
    }
  }

  float mrow[8], lrow[8];
#pragma unroll
  for (int r = 0; r < 8; ++r) { mrow[r] = -1e30f; lrow[r] = 0.0f; }
  v8f oacc[4] = {};

  const float scale = 0.125f;                      // 1/sqrt(64)
  const int nkb = (int)(qbase / 32u) + 2;          // uniform across the block

  // ---- async stage of one 32-key block into LDS buffer `buf`
  //  K tile: 4KB contiguous (32 rows x 128B) -> 256 16B chunks (waves 0,1)
  //  V tile: 64 rows x 64B strided           -> 256 16B chunks (waves 2,3)
  auto issue = [&](unsigned kb0, int buf) {
    if (wid < 2u) {
      const unsigned kbase = (unsigned)(uintptr_t)&kbuf[buf][0];
#pragma unroll
      for (unsigned ii = 0; ii < 4u; ++ii) {
        const unsigned c   = wid * 128u + ii * 32u + lane;   // 0..255
        const unsigned row = c >> 3, part = c & 7u;
        async_copy16(kbase + row * (KROW_ * 2u) + part * 16u,
                     Kb + (size_t)(kb0 + row) * HD_ + part * 8u);
      }
    } else {
      const unsigned vbase = (unsigned)(uintptr_t)&vbuf[buf][0];
#pragma unroll
      for (unsigned ii = 0; ii < 4u; ++ii) {
        const unsigned c = (wid - 2u) * 128u + ii * 32u + lane;  // 0..255
        const unsigned d = c >> 2, part = c & 3u;
        async_copy16(vbase + d * (VROW_ * 2u) + part * 16u,
                     Vtb + (size_t)d * S_ + kb0 + part * 8u);
      }
    }
  };

  issue(0, 0);

  for (int blk = 0; blk < nkb; ++blk) {
    const int cur = blk & 1;
    const unsigned kb0 = (unsigned)blk * 32u;
    wait_async0();        // my wave's DMA done
    __syncthreads();      // everyone's DMA visible
    if (blk + 1 < nkb) issue(kb0 + 32u, cur ^ 1);

    // ---- scores: S = Q K^T for two 16-key tiles (K read from LDS)
    float st[2][8];
#pragma unroll
    for (int t = 0; t < 2; ++t) {
      const _Float16* krow = &kbuf[cur][((unsigned)t * 16u + lr) * KROW_];
      h8 y0 = *(const h8*)(krow + hs * 8u);
      h8 y1 = *(const h8*)(krow + 16u + hs * 8u);
      h8 y2 = *(const h8*)(krow + 32u + hs * 8u);
      h8 y3 = *(const h8*)(krow + 48u + hs * 8u);
      v16h kf0, kf1;
#pragma unroll
      for (int e = 0; e < 8; ++e) {
        kf0[e] = y0[e]; kf0[e + 8] = y1[e];
        kf1[e] = y2[e]; kf1[e + 8] = y3[e];
      }
      v8f c = {};
      c = __builtin_amdgcn_wmma_f32_16x16x32_f16(false, qa0, false, kf0,
                                                 (short)0, c, false, false);
      c = __builtin_amdgcn_wmma_f32_16x16x32_f16(false, qa1, false, kf1,
                                                 (short)0, c, false, false);
      const int j = (int)(kb0 + (unsigned)t * 16u + lr);
      const float mj = am[j];
#pragma unroll
      for (int r = 0; r < 8; ++r) {
        const int i = (int)(qt * 16u) + r + 8 * (int)hs;
        float v = c[r] * scale + mj;
        if (j > i)            v += NEG_;   // causal (triu, diag=1)
        if (j >= i - WINDOW_) v += NEG_;   // window (triu, diag=-512)
        st[t][r] = v;
      }
    }

    // ---- online softmax update
    float pt[2][8];
#pragma unroll
    for (int r = 0; r < 8; ++r) {
      const float loc  = fmaxf(st[0][r], st[1][r]);
      const float mnew = fmaxf(mrow[r], rowmax16(loc));
      const float corr = __expf(mrow[r] - mnew);
      const float p0 = __expf(st[0][r] - mnew);
      const float p1 = __expf(st[1][r] - mnew);
      lrow[r] = lrow[r] * corr + rowsum16(p0 + p1);
      mrow[r] = mnew;
      pt[0][r] = p0; pt[1][r] = p1;
#pragma unroll
      for (int tt = 0; tt < 4; ++tt) oacc[tt][r] *= corr;
    }

    // ---- P: C-layout -> LDS -> A-layout fragment (per-wave region)
#pragma unroll
    for (int r = 0; r < 8; ++r) {
      const unsigned m = (unsigned)r + 8u * hs;
      plds[m * 32u + lr]       = (_Float16)pt[0][r];
      plds[m * 32u + 16u + lr] = (_Float16)pt[1][r];
    }
    asm volatile("s_wait_dscnt 0x0" ::: "memory");
    v16h pa;
    {
      const _Float16* pr = plds + lr * 32u;
#pragma unroll
      for (int e = 0; e < 8; ++e) {
        pa[e]     = pr[hs * 16u + e];
        pa[e + 8] = pr[hs * 16u + 8u + e];
      }
    }
    asm volatile("" ::: "memory");

    // ---- O += P V : V^T tile in LDS, lane = d col, 16 contiguous keys
#pragma unroll
    for (int tt = 0; tt < 4; ++tt) {
      const unsigned dcol = (unsigned)tt * 16u + lr;
      const _Float16* vp = &vbuf[cur][dcol * VROW_ + hs * 16u];
      h8 z0 = *(const h8*)(vp);
      h8 z1 = *(const h8*)(vp + 8);
      v16h vf;
#pragma unroll
      for (int e = 0; e < 8; ++e) { vf[e] = z0[e]; vf[e + 8] = z1[e]; }
      oacc[tt] = __builtin_amdgcn_wmma_f32_16x16x32_f16(false, pa, false, vf,
                                                        (short)0, oacc[tt],
                                                        false, false);
    }
  }

  // ---- epilogue: out[b, s, h*HD + d] = o / l
#pragma unroll
  for (int tt = 0; tt < 4; ++tt) {
#pragma unroll
    for (int r = 0; r < 8; ++r) {
      const unsigned m = (unsigned)r + 8u * hs;
      const unsigned s = qt * 16u + m;
      const unsigned col = h * HD_ + (unsigned)tt * 16u + lr;
      out[((size_t)b * S_ + s) * H_ + col] = oacc[tt][r] / lrow[r];
    }
  }
}

// ---------------------------------------------------------------------------
extern "C" void kernel_launch(void* const* d_in, const int* in_sizes, int n_in,
                              void* d_out, int out_size, void* d_ws, size_t ws_size,
                              hipStream_t stream) {
  const float* X     = (const float*)d_in[0];  // [B,S,H]
  const float* amask = (const float*)d_in[1];  // [B,1,1,S]
  const float* Wq    = (const float*)d_in[2];
  const float* bq    = (const float*)d_in[3];
  const float* Wk    = (const float*)d_in[4];
  const float* bk    = (const float*)d_in[5];
  const float* Wv    = (const float*)d_in[6];
  const float* bv    = (const float*)d_in[7];
  float* out = (float*)d_out;

  const size_t elems = (size_t)B_ * NH_ * S_ * HD_;   // 4 Mi f16 each
  _Float16* qh  = (_Float16*)d_ws;
  _Float16* kh  = qh + elems;
  _Float16* vth = kh + elems;

  // (B*S/16) * (H/64) = 4096 waves, 4 waves/block -> 1024 blocks
  const int gemm_blocks = (B_ * S_ / 16) * (H_ / 64) / 4;
  qkv_gemm_f16<0><<<gemm_blocks, 128, 0, stream>>>(X, Wq, bq, qh);
  qkv_gemm_f16<0><<<gemm_blocks, 128, 0, stream>>>(X, Wk, bk, kh);
  qkv_gemm_f16<1><<<gemm_blocks, 128, 0, stream>>>(X, Wv, bv, vth);  // V^T

  const int attn_blocks = B_ * NH_ * (S_ / 64);             // 1024
  flash_attn<<<attn_blocks, 128, 0, stream>>>(qh, kh, vth, amask, out);
}
